// MoleculeGINE_61495341744575
// MI455X (gfx1250) — matrix operand order repaired and checked by reading, not compiled, not run on hardware
//
#include <hip/hip_runtime.h>
#include <hip/hip_bf16.h>

typedef __attribute__((ext_vector_type(16))) _Float16 v16h;
typedef __attribute__((ext_vector_type(8)))  float    v8f;

// ---------------------------------------------------------------------------
// WMMA fragment helpers (CDNA5 v_wmma_f32_16x16x32_f16, wave32)
// A 16x32 f16: lane<16 -> M=lane, K: a[0..7]=0..7,  a[8..15]=16..23
//              lane>=16 -> M=lane-16, a[0..7]=8..15, a[8..15]=24..31
// B 32x16 f16 (stored transposed [N][K] in LDS): same K order as A
// C 16x16 f32: N = lane&15, M = r + 8*(lane>>4)
// ---------------------------------------------------------------------------

__device__ __forceinline__ v16h frag_a(const _Float16* __restrict__ As, int lda,
                                       int m0, int k0, int lane) {
  const int m  = m0 + (lane & 15);
  const int hf = (lane >> 4) << 3;                 // 0 or 8
  const _Float16* r = As + m * lda + k0;
  v16h a;
#pragma unroll
  for (int i = 0; i < 8; ++i) {
    a[i]     = r[hf + i];
    a[8 + i] = r[16 + hf + i];
  }
  return a;
}

// Bst layout: [Ncols][K] (transposed), contiguous K per output column
__device__ __forceinline__ v16h frag_b(const _Float16* __restrict__ Bst, int K,
                                       int k0, int n0, int lane) {
  const int n  = n0 + (lane & 15);
  const int hf = (lane >> 4) << 3;                 // 0 or 8
  const _Float16* p = Bst + n * K + k0 + hf;
  v16h b;
#pragma unroll
  for (int i = 0; i < 8; ++i) { b[i] = p[i]; b[8 + i] = p[16 + i]; }
  return b;
}

__device__ __forceinline__ v8f wmma_f16(v16h a, v16h b, v8f c) {
  return __builtin_amdgcn_wmma_f32_16x16x32_f16(false, a, false, b, (short)0, c,
                                                false, false);
}

// ---------------------------------------------------------------------------
// Prep kernels (tiny)
// ---------------------------------------------------------------------------

// Wt[c*32 + r] = (r<16) ? sum_k we_w[r,k] * lw[k,c] : 0   (f16, K padded 16->32)
__global__ void k_combine_W(const float* __restrict__ we_w,
                            const float* __restrict__ lw,
                            _Float16* __restrict__ Wt, int D) {
  int idx = blockIdx.x * 256 + threadIdx.x;
  if (idx >= 32 * D) return;
  int c = idx >> 5, r = idx & 31;
  float s = 0.f;
  if (r < 16)
    for (int k = 0; k < 128; ++k) s += we_w[r * 128 + k] * lw[k * D + c];
  Wt[idx] = (_Float16)s;
}

// be[c] = sum_k we_b[k] * lw[k,c] + lb[c]
__global__ void k_combine_be(const float* __restrict__ we_b,
                             const float* __restrict__ lw,
                             const float* __restrict__ lb,
                             float* __restrict__ be, int D) {
  int c = blockIdx.x * 256 + threadIdx.x;
  if (c >= D) return;
  float s = lb[c];
  for (int k = 0; k < 128; ++k) s += we_b[k] * lw[k * D + c];
  be[c] = s;
}

__global__ void k_cvt16(const float* __restrict__ src, _Float16* __restrict__ dst,
                        int n) {
  int i = blockIdx.x * 256 + threadIdx.x;
  if (i < n) dst[i] = (_Float16)src[i];
}

// src [K][Ncols] f32 row-major -> dst [Ncols][K] f16 (transposed for frag_b)
__global__ void k_cvt16_t(const float* __restrict__ src, _Float16* __restrict__ dst,
                          int K, int Ncols) {
  int idx = blockIdx.x * 256 + threadIdx.x;
  if (idx >= K * Ncols) return;
  int k = idx / Ncols, n = idx % Ncols;
  dst[n * K + k] = (_Float16)src[idx];
}

__global__ void k_bn_fold(const float* __restrict__ g, const float* __restrict__ b,
                          const float* __restrict__ m, const float* __restrict__ v,
                          float* __restrict__ s, float* __restrict__ t) {
  int c = threadIdx.x;
  float sc = g[c] * rsqrtf(v[c] + 1e-5f);
  s[c] = sc;
  t[c] = b[c] - m[c] * sc;
}

// cls_w [128][10] -> dst [16][128] f16, cols 10..15 zero
__global__ void k_cls_pad(const float* __restrict__ wc, _Float16* __restrict__ dst) {
  int idx = blockIdx.x * 256 + threadIdx.x;
  if (idx >= 16 * 128) return;
  int n = idx >> 7, k = idx & 127;
  dst[idx] = (n < 10) ? (_Float16)wc[k * 10 + n] : (_Float16)0.f;
}

// ---------------------------------------------------------------------------
// Fused edge kernel: e = edge_attr16 @ Wt + be; msg = relu(x[src] + e);
// atomicAdd into aggr[dst].  64 edges per block, 16 per wave, K=32 (padded 16).
// ---------------------------------------------------------------------------
template <int D>
__global__ __launch_bounds__(128) void k_edge(
    const _Float16* __restrict__ ea16,  // [E][16]
    const _Float16* __restrict__ Wt,    // [D][32] f16 (transposed, K padded)
    const float* __restrict__ be,       // [D]
    const float* __restrict__ x,        // [N][D]
    const int* __restrict__ src, const int* __restrict__ dst,
    float* __restrict__ aggr,           // [N][D]
    int E_) {
  __shared__ _Float16 As[64 * 32];
  __shared__ _Float16 Bs[D * 32];
  __shared__ int s_src[64], s_dst[64];
  const int tid = threadIdx.x;
  const int e0 = blockIdx.x * 64;
  const bool full = (e0 + 64 <= E_);   // block-uniform fast path
  for (int i = tid; i < D * 32; i += 128) Bs[i] = Wt[i];
  if (full) {
    for (int i = tid; i < 64 * 32; i += 128) {
      int r = i >> 5, c = i & 31;
      As[i] = (c < 16) ? ea16[(e0 + r) * 16 + c] : (_Float16)0.f;
    }
    if (tid < 64) {
      s_src[tid] = src[e0 + tid];
      s_dst[tid] = dst[e0 + tid];
    }
  } else {
    for (int i = tid; i < 64 * 32; i += 128) {
      int r = i >> 5, c = i & 31;
      int e = e0 + r;
      As[i] = (c < 16 && e < E_) ? ea16[e * 16 + c] : (_Float16)0.f;
    }
    if (tid < 64) {
      int e = e0 + tid;
      s_src[tid] = (e < E_) ? src[e] : 0;
      s_dst[tid] = (e < E_) ? dst[e] : 0;
    }
  }
  __syncthreads();

  const int wave = tid >> 5, lane = tid & 31;
  const int m0 = wave * 16;
  const int nl = lane & 15;
  const int mh = (lane >> 4) * 8;
  v16h a = frag_a(As, 32, m0, 0, lane);
#pragma unroll
  for (int nt = 0; nt < D / 16; ++nt) {
    v16h b = frag_b(Bs, 32, 0, nt * 16, lane);
    v8f acc = {};
    acc = wmma_f16(a, b, acc);
    const int n = nt * 16 + nl;
    const float bias = be[n];
    if (full) {
#pragma unroll
      for (int r = 0; r < 8; ++r) {
        const int le = m0 + r + mh;
        const int s = s_src[le], d = s_dst[le];
        float v = acc[r] + bias + x[s * D + n];
        v = v > 0.f ? v : 0.f;
        atomicAdd(&aggr[d * D + n], v);
      }
    } else {
#pragma unroll
      for (int r = 0; r < 8; ++r) {
        const int le = m0 + r + mh;
        if (e0 + le < E_) {
          const int s = s_src[le], d = s_dst[le];
          float v = acc[r] + bias + x[s * D + n];
          v = v > 0.f ? v : 0.f;
          atomicAdd(&aggr[d * D + n], v);
        }
      }
    }
  }
}

// ---------------------------------------------------------------------------
// Node MLP stage A: hid = relu((x + aggr) @ w1 + b1), stored f16. [N,K]->[N,128]
// ---------------------------------------------------------------------------
template <int K>
__global__ __launch_bounds__(128) void k_mlp_a(
    const float* __restrict__ x, const float* __restrict__ aggr,
    const _Float16* __restrict__ w1t,  // [128][K] f16 transposed
    const float* __restrict__ b1,
    _Float16* __restrict__ hid,        // [N][128] f16
    int Nn) {
  __shared__ _Float16 As[64 * K];
  __shared__ _Float16 Bs[128 * K];
  const int tid = threadIdx.x;
  const int r0 = blockIdx.x * 64;
  const bool full = (r0 + 64 <= Nn);
  for (int i = tid; i < 128 * K; i += 128) Bs[i] = w1t[i];
  if (full) {
    for (int i = tid; i < 64 * K; i += 128) {
      int r = i / K, c = i % K;
      int row = r0 + r;
      As[i] = (_Float16)(x[row * K + c] + aggr[row * K + c]);
    }
  } else {
    for (int i = tid; i < 64 * K; i += 128) {
      int r = i / K, c = i % K;
      int row = r0 + r;
      As[i] = (row < Nn) ? (_Float16)(x[row * K + c] + aggr[row * K + c])
                         : (_Float16)0.f;
    }
  }
  __syncthreads();

  const int wave = tid >> 5, lane = tid & 31;
  const int m0 = wave * 16;
  v16h a[K / 32];
#pragma unroll
  for (int ks = 0; ks < K / 32; ++ks) a[ks] = frag_a(As, K, m0, ks * 32, lane);
  const int nl = lane & 15, mh = (lane >> 4) * 8;
#pragma unroll
  for (int nt = 0; nt < 8; ++nt) {
    v8f acc = {};
#pragma unroll
    for (int ks = 0; ks < K / 32; ++ks) {
      v16h b = frag_b(Bs, K, ks * 32, nt * 16, lane);
      acc = wmma_f16(a[ks], b, acc);
    }
    const int n = nt * 16 + nl;
    const float bias = b1[n];
    if (full) {
#pragma unroll
      for (int r = 0; r < 8; ++r) {
        int row = r0 + m0 + r + mh;
        float v = acc[r] + bias;
        v = v > 0.f ? v : 0.f;
        hid[row * 128 + n] = (_Float16)v;
      }
    } else {
#pragma unroll
      for (int r = 0; r < 8; ++r) {
        int row = r0 + m0 + r + mh;
        if (row < Nn) {
          float v = acc[r] + bias;
          v = v > 0.f ? v : 0.f;
          hid[row * 128 + n] = (_Float16)v;
        }
      }
    }
  }
}

// ---------------------------------------------------------------------------
// Node MLP stage B + folded BN + relu: h = relu((hid @ w2 + b2) * s + t)  f32
// ---------------------------------------------------------------------------
__global__ __launch_bounds__(128) void k_mlp_b(
    const _Float16* __restrict__ hid,   // [N][128]
    const _Float16* __restrict__ w2t,   // [128][128] transposed
    const float* __restrict__ b2, const float* __restrict__ bns,
    const float* __restrict__ bnt, float* __restrict__ h, int Nn) {
  __shared__ _Float16 As[64 * 128];
  __shared__ _Float16 Bs[128 * 128];
  const int tid = threadIdx.x;
  const int r0 = blockIdx.x * 64;
  const bool full = (r0 + 64 <= Nn);
  for (int i = tid; i < 128 * 128; i += 128) Bs[i] = w2t[i];
  if (full) {
    for (int i = tid; i < 64 * 128; i += 128) {
      int r = i >> 7, c = i & 127;
      As[i] = hid[(r0 + r) * 128 + c];
    }
  } else {
    for (int i = tid; i < 64 * 128; i += 128) {
      int r = i >> 7, c = i & 127;
      int row = r0 + r;
      As[i] = (row < Nn) ? hid[row * 128 + c] : (_Float16)0.f;
    }
  }
  __syncthreads();

  const int wave = tid >> 5, lane = tid & 31;
  const int m0 = wave * 16;
  v16h a[4];
#pragma unroll
  for (int ks = 0; ks < 4; ++ks) a[ks] = frag_a(As, 128, m0, ks * 32, lane);
  const int nl = lane & 15, mh = (lane >> 4) * 8;
#pragma unroll
  for (int nt = 0; nt < 8; ++nt) {
    v8f acc = {};
#pragma unroll
    for (int ks = 0; ks < 4; ++ks) {
      v16h b = frag_b(Bs, 128, ks * 32, nt * 16, lane);
      acc = wmma_f16(a[ks], b, acc);
    }
    const int n = nt * 16 + nl;
    const float bias = b2[n], sc = bns[n], sh = bnt[n];
    if (full) {
#pragma unroll
      for (int r = 0; r < 8; ++r) {
        int row = r0 + m0 + r + mh;
        float v = (acc[r] + bias) * sc + sh;
        v = v > 0.f ? v : 0.f;
        h[row * 128 + n] = v;
      }
    } else {
#pragma unroll
      for (int r = 0; r < 8; ++r) {
        int row = r0 + m0 + r + mh;
        if (row < Nn) {
          float v = (acc[r] + bias) * sc + sh;
          v = v > 0.f ? v : 0.f;
          h[row * 128 + n] = v;
        }
      }
    }
  }
}

// ---------------------------------------------------------------------------
// Global add pool: pooled[batch[n]] += h[n]
// ---------------------------------------------------------------------------
__global__ void k_pool(const float* __restrict__ h, const int* __restrict__ batch,
                       float* __restrict__ pooled) {
  int node = blockIdx.x, c = threadIdx.x;
  atomicAdd(&pooled[batch[node] * 128 + c], h[node * 128 + c]);
}

// ---------------------------------------------------------------------------
// Classifier: out[G,10] = pooled @ cls_w + cls_b  (N padded to 16)
// ---------------------------------------------------------------------------
__global__ __launch_bounds__(128) void k_cls(
    const float* __restrict__ pooled,     // [G][128]
    const _Float16* __restrict__ wc16,    // [16][128] transposed, padded
    const float* __restrict__ bc, float* __restrict__ out, int Gn) {
  __shared__ _Float16 As[64 * 128];
  __shared__ _Float16 Bs[16 * 128];
  const int tid = threadIdx.x;
  const int r0 = blockIdx.x * 64;
  const bool full = (r0 + 64 <= Gn);
  for (int i = tid; i < 16 * 128; i += 128) Bs[i] = wc16[i];
  if (full) {
    for (int i = tid; i < 64 * 128; i += 128) {
      int r = i >> 7, c = i & 127;
      As[i] = (_Float16)pooled[(r0 + r) * 128 + c];
    }
  } else {
    for (int i = tid; i < 64 * 128; i += 128) {
      int r = i >> 7, c = i & 127;
      int row = r0 + r;
      As[i] = (row < Gn) ? (_Float16)pooled[row * 128 + c] : (_Float16)0.f;
    }
  }
  __syncthreads();

  const int wave = tid >> 5, lane = tid & 31;
  const int m0 = wave * 16;
  v8f acc = {};
#pragma unroll
  for (int ks = 0; ks < 4; ++ks) {
    v16h a = frag_a(As, 128, m0, ks * 32, lane);
    v16h b = frag_b(Bs, 128, ks * 32, 0, lane);
    acc = wmma_f16(a, b, acc);
  }
  const int n = lane & 15;
  const int mh = (lane >> 4) * 8;
  if (n < 10) {
    const float bias = bc[n];
    if (full) {
#pragma unroll
      for (int r = 0; r < 8; ++r) {
        int row = r0 + m0 + r + mh;
        out[row * 10 + n] = acc[r] + bias;
      }
    } else {
#pragma unroll
      for (int r = 0; r < 8; ++r) {
        int row = r0 + m0 + r + mh;
        if (row < Gn) out[row * 10 + n] = acc[r] + bias;
      }
    }
  }
}

// ---------------------------------------------------------------------------
// Host launcher
// ---------------------------------------------------------------------------
extern "C" void kernel_launch(void* const* d_in, const int* in_sizes, int n_in,
                              void* d_out, int out_size, void* d_ws, size_t ws_size,
                              hipStream_t stream) {
  const int Nn = in_sizes[0] / 32;       // 100000
  const int Ee = in_sizes[2] / 16;       // 1600000
  const int Gn = out_size / 10;          // 2048

  const float* x     = (const float*)d_in[0];
  const int*   ei    = (const int*)d_in[1];
  const float* ea    = (const float*)d_in[2];
  const int*   batch = (const int*)d_in[3];
  const float* we_w  = (const float*)d_in[4];
  const float* we_b  = (const float*)d_in[5];
  const float *lw[3], *lb[3], *w1[3], *b1[3], *w2[3], *b2[3];
  const float *bng[3], *bnb[3], *bnm[3], *bnv[3];
  for (int i = 0; i < 3; ++i) {
    int b = 6 + i * 10;
    lw[i]  = (const float*)d_in[b + 0]; lb[i]  = (const float*)d_in[b + 1];
    w1[i]  = (const float*)d_in[b + 2]; b1[i]  = (const float*)d_in[b + 3];
    w2[i]  = (const float*)d_in[b + 4]; b2[i]  = (const float*)d_in[b + 5];
    bng[i] = (const float*)d_in[b + 6]; bnb[i] = (const float*)d_in[b + 7];
    bnm[i] = (const float*)d_in[b + 8]; bnv[i] = (const float*)d_in[b + 9];
  }
  const float* cls_w = (const float*)d_in[36];
  const float* cls_b = (const float*)d_in[37];
  const int* src = ei;
  const int* dst = ei + Ee;

  // Workspace carve (deterministic every call)
  char* p = (char*)d_ws;
  auto carve = [&](size_t bytes) -> void* {
    void* r = (void*)p;
    p += (bytes + 255) & ~(size_t)255;
    return r;
  };
  _Float16* ea16 = (_Float16*)carve((size_t)Ee * 16 * 2);
  const int Dd[3] = {32, 128, 128};
  _Float16 *Wt[3], *w1t[3], *w2t[3];
  float *be[3], *bns[3], *bnt[3];
  for (int i = 0; i < 3; ++i) {
    Wt[i]  = (_Float16*)carve((size_t)Dd[i] * 32 * 2);
    be[i]  = (float*)carve((size_t)Dd[i] * 4);
    w1t[i] = (_Float16*)carve((size_t)128 * Dd[i] * 2);
    w2t[i] = (_Float16*)carve((size_t)128 * 128 * 2);
    bns[i] = (float*)carve(128 * 4);
    bnt[i] = (float*)carve(128 * 4);
  }
  _Float16* cls16 = (_Float16*)carve((size_t)16 * 128 * 2);
  float* aggr     = (float*)carve((size_t)Nn * 128 * 4);
  _Float16* hid   = (_Float16*)carve((size_t)Nn * 128 * 2);
  float* h        = (float*)carve((size_t)Nn * 128 * 4);
  float* pooled   = (float*)carve((size_t)Gn * 128 * 4);

  // --- prep ---
  k_cvt16<<<(Ee * 16 + 255) / 256, 256, 0, stream>>>(ea, ea16, Ee * 16);
  for (int i = 0; i < 3; ++i) {
    int D = Dd[i];
    k_combine_W<<<(32 * D + 255) / 256, 256, 0, stream>>>(we_w, lw[i], Wt[i], D);
    k_combine_be<<<(D + 255) / 256, 256, 0, stream>>>(we_b, lw[i], lb[i], be[i], D);
    k_cvt16_t<<<(D * 128 + 255) / 256, 256, 0, stream>>>(w1[i], w1t[i], D, 128);
    k_cvt16_t<<<(128 * 128 + 255) / 256, 256, 0, stream>>>(w2[i], w2t[i], 128, 128);
    k_bn_fold<<<1, 128, 0, stream>>>(bng[i], bnb[i], bnm[i], bnv[i], bns[i], bnt[i]);
  }
  k_cls_pad<<<(16 * 128 + 255) / 256, 256, 0, stream>>>(cls_w, cls16);

  const int gridE = (Ee + 63) / 64;
  const int gridN = (Nn + 63) / 64;

  // --- layer 1 (d_in = 32) ---
  hipMemsetAsync(aggr, 0, (size_t)Nn * 32 * 4, stream);
  k_edge<32><<<gridE, 128, 0, stream>>>(ea16, Wt[0], be[0], x, src, dst, aggr, Ee);
  k_mlp_a<32><<<gridN, 128, 0, stream>>>(x, aggr, w1t[0], b1[0], hid, Nn);
  k_mlp_b<<<gridN, 128, 0, stream>>>(hid, w2t[0], b2[0], bns[0], bnt[0], h, Nn);

  // --- layer 2 ---
  hipMemsetAsync(aggr, 0, (size_t)Nn * 128 * 4, stream);
  k_edge<128><<<gridE, 128, 0, stream>>>(ea16, Wt[1], be[1], h, src, dst, aggr, Ee);
  k_mlp_a<128><<<gridN, 128, 0, stream>>>(h, aggr, w1t[1], b1[1], hid, Nn);
  k_mlp_b<<<gridN, 128, 0, stream>>>(hid, w2t[1], b2[1], bns[1], bnt[1], h, Nn);

  // --- layer 3 ---
  hipMemsetAsync(aggr, 0, (size_t)Nn * 128 * 4, stream);
  k_edge<128><<<gridE, 128, 0, stream>>>(ea16, Wt[2], be[2], h, src, dst, aggr, Ee);
  k_mlp_a<128><<<gridN, 128, 0, stream>>>(h, aggr, w1t[2], b1[2], hid, Nn);
  k_mlp_b<<<gridN, 128, 0, stream>>>(hid, w2t[2], b2[2], bns[2], bnt[2], h, Nn);

  // --- pool + classifier ---
  hipMemsetAsync(pooled, 0, (size_t)Gn * 128 * 4, stream);
  k_pool<<<Nn, 128, 0, stream>>>(h, batch, pooled);
  k_cls<<<(Gn + 63) / 64, 128, 0, stream>>>(pooled, cls16, cls_b, (float*)d_out, Gn);
}
